// Attention_38869454029450
// MI455X (gfx1250) — compile-verified, matrix operand order checked by reading, not run on hardware
//
#include <hip/hip_runtime.h>
#include <hip/hip_bf16.h>

// ---------------------------------------------------------------------------
// Fused attention block for MI455X (gfx1250, wave32, WMMA f16 -> f32 acc)
//   out = (softmax(rope(rms(XWq)) rope(rms(XWk))^T * 1/sqrt(D)) (XWv)) Wout
// B=2 N=4096 C=1024 H=16 D=64.  Compute-bound (~206 GFLOP vs ~6us of HBM at
// 23.3 TB/s), so everything runs on v_wmma_f32_16x16x32_f16 with f32 softmax.
// ---------------------------------------------------------------------------

typedef __attribute__((ext_vector_type(16))) _Float16 v16h;
typedef __attribute__((ext_vector_type(8)))  _Float16 v8h;
typedef __attribute__((ext_vector_type(8)))  float    v8f;

#define B_   2
#define N_   4096
#define C_   1024
#define H_   16
#define D_   64
#define BN_  8192
#define C3_  3072

// ---------------------------------------------------------------------------
// gfx1250 async global->LDS copy (ASYNCcnt-tracked). The builtin takes
// (global int4* src, local int4* dst, int offset, int cpol) — int4 in GCC
// vector_size(16) form per the compiler's own diagnostic.
// ---------------------------------------------------------------------------
#if defined(__has_builtin)
#if __has_builtin(__builtin_amdgcn_global_load_async_to_lds_b128)
#define USE_ASYNC_LDS 1
#endif
#endif

typedef int b128_t __attribute__((vector_size(16)));
typedef __attribute__((address_space(1))) b128_t* gb128_p;
typedef __attribute__((address_space(3))) b128_t* lb128_p;

static __device__ __forceinline__ void cp_b128(const _Float16* g, _Float16* l) {
#ifdef USE_ASYNC_LDS
  __builtin_amdgcn_global_load_async_to_lds_b128(
      (gb128_p)(unsigned long long)(uintptr_t)g,
      (lb128_p)(unsigned)(uintptr_t)l,        // generic LDS addr: low 32 bits
      0, 0);
#else
  *reinterpret_cast<v8h*>(l) = *reinterpret_cast<const v8h*>(g);
#endif
}

static __device__ __forceinline__ void async_wait() {
#ifdef USE_ASYNC_LDS
#if __has_builtin(__builtin_amdgcn_s_wait_asynccnt)
  __builtin_amdgcn_s_wait_asynccnt(0);
#else
  asm volatile("s_wait_asynccnt 0x0" ::: "memory");
#endif
#endif
}

// ---------------------------------------------------------------------------
// Fragment loader (ISA 7.12.2 layouts).
// A-frag (16xK): lane l holds row (l&15); K chunks at (l>>4)*8 and 16+(l>>4)*8.
// B-frag (Kx16): lane l holds col (l&15) with the same K split.
// `stride` is always a multiple of 8 halves => 16B-aligned v8h accesses.
// ---------------------------------------------------------------------------
static __device__ __forceinline__ v16h frag_ld(const _Float16* base, int stride) {
  const int lane = threadIdx.x & 31;
  const _Float16* p = base + (lane & 15) * stride + ((lane >> 4) << 3);
  v8h lo = *reinterpret_cast<const v8h*>(p);
  v8h hi = *reinterpret_cast<const v8h*>(p + 16);
  return __builtin_shufflevector(lo, hi, 0, 1, 2, 3, 4, 5, 6, 7,
                                 8, 9, 10, 11, 12, 13, 14, 15);
}

static __device__ __forceinline__ v8f wmma_f16(v16h a, v16h b, v8f c) {
  return __builtin_amdgcn_wmma_f32_16x16x32_f16(false, a, false, b,
                                                (short)0, c, false, false);
}

static __device__ __forceinline__ v8f zero_v8f() {
  v8f z = {0.f, 0.f, 0.f, 0.f, 0.f, 0.f, 0.f, 0.f};
  return z;
}

// ---------------------------------------------------------------------------
// fp32 -> f16 elementwise, and transposed f16 weight copies
// ---------------------------------------------------------------------------
__global__ __launch_bounds__(256) void f32_to_f16_kernel(
    const float* __restrict__ in, _Float16* __restrict__ out, int n) {
  int i = blockIdx.x * 256 + threadIdx.x;
  if (i < n) out[i] = (_Float16)in[i];
}

__global__ __launch_bounds__(256) void transpose_f32_to_f16_kernel(
    const float* __restrict__ W, _Float16* __restrict__ WT, int K, int Ncol) {
  int i = blockIdx.x * 256 + threadIdx.x;
  if (i < K * Ncol) {
    int k = i / Ncol;
    int n = i - k * Ncol;
    WT[(size_t)n * K + k] = (_Float16)W[i];
  }
}

// ---------------------------------------------------------------------------
// GEMM: out[M][Nn] = A[M][Kk] * BT[Nn][Kk]^T   (f16 in, f32 accumulate)
// 256 threads = 8 waves (4x2), block tile 128x128, wave tile 32x64
// (2x4 fragments -> 8 wmma per wave per 32-deep K step).
// LDS ping-pong double buffering; stage for tile k+1 is issued (async when
// available) before computing tile k.
// MODE 0: scatter f16 into qkv3 [3][B][H][N][D].   MODE 1: fp32 row-major.
// ---------------------------------------------------------------------------
template <int MODE>
__global__ __launch_bounds__(256) void gemm_f16_wmma(
    const _Float16* __restrict__ A, const _Float16* __restrict__ BT,
    void* __restrict__ out, int M, int Nn, int Kk) {
  __shared__ _Float16 Alds[2][128 * 40];   // 32 k-halves + 8 pad
  __shared__ _Float16 Blds[2][128 * 40];

  const int lane = threadIdx.x & 31;
  const int l16  = lane & 15;
  const int wid  = threadIdx.x >> 5;
  const int wrow = wid >> 1;               // 0..3  -> rows wrow*32
  const int wcol = wid & 1;                // 0..1  -> cols wcol*64
  const int m0 = blockIdx.y * 128;
  const int n0 = blockIdx.x * 128;

  v8f acc[2][4];
#pragma unroll
  for (int i = 0; i < 2; ++i)
#pragma unroll
    for (int j = 0; j < 4; ++j) acc[i][j] = zero_v8f();

  auto stage = [&](int k0, int buf) {
    for (int t = threadIdx.x; t < 512; t += 256) {
      int r = t >> 2, c = t & 3;
      cp_b128(A + (size_t)(m0 + r) * Kk + k0 + c * 8, &Alds[buf][r * 40 + c * 8]);
    }
    for (int t = threadIdx.x; t < 512; t += 256) {
      int r = t >> 2, c = t & 3;
      cp_b128(BT + (size_t)(n0 + r) * Kk + k0 + c * 8, &Blds[buf][r * 40 + c * 8]);
    }
  };

  const int nk = Kk >> 5;                  // K steps of 32
  stage(0, 0);
  for (int kk = 0; kk < nk; ++kk) {
    async_wait();
    __syncthreads();
    if (kk + 1 < nk) stage((kk + 1) << 5, (kk + 1) & 1);

    const _Float16* Ab = Alds[kk & 1];
    const _Float16* Bb = Blds[kk & 1];
    v16h af[2], bf[4];
#pragma unroll
    for (int i = 0; i < 2; ++i)
      af[i] = frag_ld(&Ab[(wrow * 32 + i * 16) * 40], 40);
#pragma unroll
    for (int j = 0; j < 4; ++j)
      bf[j] = frag_ld(&Bb[(wcol * 64 + j * 16) * 40], 40);
#pragma unroll
    for (int i = 0; i < 2; ++i)
#pragma unroll
      for (int j = 0; j < 4; ++j) acc[i][j] = wmma_f16(af[i], bf[j], acc[i][j]);
  }

  // epilogue: C/D layout -> element (m = r + 8*(lane>>4), n = lane&15)
#pragma unroll
  for (int i = 0; i < 2; ++i) {
#pragma unroll
    for (int j = 0; j < 4; ++j) {
#pragma unroll
      for (int r = 0; r < 8; ++r) {
        int row = m0 + wrow * 32 + i * 16 + r + ((lane >> 4) << 3);
        int col = n0 + wcol * 64 + j * 16 + l16;
        float val = acc[i][j][r];
        if (MODE == 0) {
          int s = col >> 10, rem = col & 1023;
          int hh = rem >> 6, d = rem & 63;
          int bb = row >> 12, nn = row & 4095;
          ((_Float16*)out)[((((size_t)s * B_ + bb) * H_ + hh) * N_ + nn) * D_ + d] =
              (_Float16)val;
        } else {
          ((float*)out)[(size_t)row * Nn + col] = val;
        }
      }
    }
  }
}

// ---------------------------------------------------------------------------
// Per-(s,b,h,n) RMSNorm + RoPE, in place on qkv3 f16 (s=0:q, s=1:k).
// One wave per D=64 row; lane holds d=lane and d=lane+32 (rotate_half pair).
// Folds the attention scale AND log2(e) into Q so attention can use exp2.
// ---------------------------------------------------------------------------
__global__ __launch_bounds__(128) void rmsnorm_rope_kernel(
    _Float16* __restrict__ qkv3, const float* __restrict__ cosp,
    const float* __restrict__ sinp, const float* __restrict__ gq,
    const float* __restrict__ gk) {
  const int lane = threadIdx.x & 31;
  const int wid  = threadIdx.x >> 5;
  int row = blockIdx.x * 4 + wid;          // 0 .. 2*B*H*N-1
  int s = row >> 17;                       // / (B*H*N = 131072)
  int rem = row & 131071;
  int b = rem >> 16;                       // / (H*N = 65536)
  rem &= 65535;
  int h = rem >> 12;                       // / N
  int n = rem & 4095;

  _Float16* p = qkv3 + ((((size_t)s * B_ + b) * H_ + h) * N_ + n) * D_;
  float e0 = (float)p[lane];
  float e1 = (float)p[lane + 32];

  float ss = e0 * e0 + e1 * e1;
  ss += __shfl_xor(ss, 1);
  ss += __shfl_xor(ss, 2);
  ss += __shfl_xor(ss, 4);
  ss += __shfl_xor(ss, 8);
  ss += __shfl_xor(ss, 16);
  float rinv = rsqrtf(ss * (1.0f / 64.0f) + 1e-6f);

  const float* g = (s == 0) ? gq : gk;
  float x0 = e0 * rinv * g[lane];
  float x1 = e1 * rinv * g[lane + 32];

  const float* cr = cosp + n * 64;
  const float* sr = sinp + n * 64;
  float y0 = x0 * cr[lane]      - x1 * sr[lane];        // d < 32: -x2*sin
  float y1 = x1 * cr[lane + 32] + x0 * sr[lane + 32];   // d >= 32: +x1*sin
  if (s == 0) {                    // 1/sqrt(D) * log2(e) folded into Q
    const float qs = 0.125f * 1.44269504f;
    y0 *= qs; y1 *= qs;
  }
  p[lane]      = (_Float16)y0;
  p[lane + 32] = (_Float16)y1;
}

// ---------------------------------------------------------------------------
// Flash attention. Block = 256 threads = 8 waves; block owns 128 query rows of
// one (b,h); each wave owns 16 rows with Q resident as 2 A-fragments.
// 64 keys staged per barrier pair (K async row-major; V transposed manually),
// processed as two 32-key WMMA sub-steps. Online softmax in f32 (base 2).
// Next tile prefetched with global_prefetch.
// ---------------------------------------------------------------------------
__global__ __launch_bounds__(256) void flash_attn_kernel(
    const _Float16* __restrict__ qkv3, _Float16* __restrict__ attn_out) {
  __shared__ _Float16 Klds[64 * 72];       // [key][d], stride 72
  __shared__ _Float16 Vtlds[64 * 72];      // [d][key 0..63 + pad], stride 72
  __shared__ _Float16 Plds[8 * 16 * 40];   // per-wave P patch [row][key]

  const int lane = threadIdx.x & 31;
  const int l16  = lane & 15;
  const int hlf8 = (lane >> 4) << 3;
  const int wid  = threadIdx.x >> 5;

  const int b = blockIdx.y >> 4;
  const int h = blockIdx.y & 15;
  const size_t plane = (size_t)N_ * D_;
  const _Float16* Q = qkv3 + ((size_t)((0 * B_ + b) * H_ + h)) * plane;
  const _Float16* K = qkv3 + ((size_t)((1 * B_ + b) * H_ + h)) * plane;
  const _Float16* V = qkv3 + ((size_t)((2 * B_ + b) * H_ + h)) * plane;

  const int qbase = blockIdx.x * 128 + wid * 16;

  v16h qf[2];
  qf[0] = frag_ld(Q + (size_t)qbase * D_ + 0,  D_);
  qf[1] = frag_ld(Q + (size_t)qbase * D_ + 32, D_);

  v8f o[4];
#pragma unroll
  for (int dt = 0; dt < 4; ++dt) o[dt] = zero_v8f();
  float mrow[8], lrow[8];
#pragma unroll
  for (int r = 0; r < 8; ++r) { mrow[r] = -3.0e38f; lrow[r] = 0.f; }

  _Float16* Pw = &Plds[wid * 16 * 40];

  const int ntiles = N_ / 64;
  for (int kt = 0; kt < ntiles; ++kt) {
    const int k0 = kt * 64;

    // prefetch next 64-key K/V tile (one 128B line per thread)
    if (kt + 1 < ntiles) {
      int t = threadIdx.x;
      if (t < 64)
        __builtin_prefetch((const void*)(K + (size_t)(k0 + 64 + t) * D_), 0, 1);
      else if (t < 128)
        __builtin_prefetch((const void*)(V + (size_t)(k0 + t) * D_), 0, 1);
    }

    // stage: K row-major (async), V transposed (manual)
    for (int t = threadIdx.x; t < 512; t += 256) {
      int key = t >> 3, c = t & 7;
      cp_b128(K + (size_t)(k0 + key) * D_ + c * 8, &Klds[key * 72 + c * 8]);
      v8h vv = *reinterpret_cast<const v8h*>(V + (size_t)(k0 + key) * D_ + c * 8);
#pragma unroll
      for (int i = 0; i < 8; ++i) Vtlds[(c * 8 + i) * 72 + key] = vv[i];
    }
    async_wait();
    __syncthreads();

#pragma unroll
    for (int ss = 0; ss < 2; ++ss) {       // two 32-key sub-steps
      const _Float16* Kb = &Klds[(ss * 32) * 72];

      v8f s0 = zero_v8f(), s1 = zero_v8f();
      s0 = wmma_f16(qf[0], frag_ld(&Kb[0 * 16 * 72 + 0],  72), s0);
      s0 = wmma_f16(qf[1], frag_ld(&Kb[0 * 16 * 72 + 32], 72), s0);
      s1 = wmma_f16(qf[0], frag_ld(&Kb[1 * 16 * 72 + 0],  72), s1);
      s1 = wmma_f16(qf[1], frag_ld(&Kb[1 * 16 * 72 + 32], 72), s1);

      // online softmax, base-2 (log2e folded into Q)
#pragma unroll
      for (int r = 0; r < 8; ++r) {
        float a = s0[r], c = s1[r];
        float mx = fmaxf(a, c);
        mx = fmaxf(mx, __shfl_xor(mx, 1));
        mx = fmaxf(mx, __shfl_xor(mx, 2));
        mx = fmaxf(mx, __shfl_xor(mx, 4));
        mx = fmaxf(mx, __shfl_xor(mx, 8));
        float mnew = fmaxf(mrow[r], mx);
        float alpha = exp2f(mrow[r] - mnew);
        float p0 = exp2f(a - mnew);
        float p1 = exp2f(c - mnew);
        float ps = p0 + p1;
        ps += __shfl_xor(ps, 1);
        ps += __shfl_xor(ps, 2);
        ps += __shfl_xor(ps, 4);
        ps += __shfl_xor(ps, 8);
        lrow[r] = lrow[r] * alpha + ps;
        mrow[r] = mnew;
#pragma unroll
        for (int dt = 0; dt < 4; ++dt) o[dt][r] *= alpha;
        int prow = r + hlf8;
        Pw[prow * 40 + 0 * 16 + l16] = (_Float16)p0;
        Pw[prow * 40 + 1 * 16 + l16] = (_Float16)p1;
      }

      // O(16x64) += P(16x32) * V(32x64) (same-wave LDS ops are in order)
      v16h pf = frag_ld(Pw, 40);
#pragma unroll
      for (int dt = 0; dt < 4; ++dt)
        o[dt] = wmma_f16(pf, frag_ld(&Vtlds[dt * 16 * 72 + ss * 32], 72), o[dt]);
    }
    __syncthreads();
  }

  // normalize and store f16 [B][N][C] with head offset
  const int colb = h * D_;
#pragma unroll
  for (int dt = 0; dt < 4; ++dt) {
#pragma unroll
    for (int r = 0; r < 8; ++r) {
      int nrow = qbase + r + hlf8;
      float val = o[dt][r] / lrow[r];
      attn_out[(size_t)(b * N_ + nrow) * C_ + colb + dt * 16 + l16] =
          (_Float16)val;
    }
  }
}

// ---------------------------------------------------------------------------
// Host-side launch pipeline
// ---------------------------------------------------------------------------
extern "C" void kernel_launch(void* const* d_in, const int* in_sizes, int n_in,
                              void* d_out, int out_size, void* d_ws, size_t ws_size,
                              hipStream_t stream) {
  const float* hidden = (const float*)d_in[0];
  const float* cosp   = (const float*)d_in[1];
  const float* sinp   = (const float*)d_in[2];
  const float* Wqkv   = (const float*)d_in[3];
  const float* Wout   = (const float*)d_in[4];
  const float* gq     = (const float*)d_in[5];
  const float* gk     = (const float*)d_in[6];

  // workspace layout (bytes):
  //   Xh      f16 [8192,1024]           16,777,216
  //   WqkvT   f16 [3072,1024]            6,291,456
  //   WoutT   f16 [1024,1024]            2,097,152
  //   qkv3    f16 [3][2][16][4096][64]  50,331,648
  //   attnH   f16 [8192,1024]           16,777,216   (total ~92.3 MB)
  char* ws = (char*)d_ws;
  _Float16* Xh    = (_Float16*)(ws);
  _Float16* WqkvT = (_Float16*)(ws + 16777216);
  _Float16* WoutT = (_Float16*)(ws + 16777216 + 6291456);
  _Float16* qkv3  = (_Float16*)(ws + 16777216 + 6291456 + 2097152);
  _Float16* attnH = (_Float16*)(ws + 16777216 + 6291456 + 2097152 + 50331648);

  f32_to_f16_kernel<<<dim3((BN_ * C_) / 256), 256, 0, stream>>>(
      hidden, Xh, BN_ * C_);
  transpose_f32_to_f16_kernel<<<dim3((C_ * C3_) / 256), 256, 0, stream>>>(
      Wqkv, WqkvT, C_, C3_);
  transpose_f32_to_f16_kernel<<<dim3((C_ * C_) / 256), 256, 0, stream>>>(
      Wout, WoutT, C_, C_);

  // QKV projection, scattered into [3][B][H][N][D] f16
  gemm_f16_wmma<0><<<dim3(C3_ / 128, BN_ / 128), 256, 0, stream>>>(
      Xh, WqkvT, (void*)qkv3, BN_, C3_, C_);

  // per-head RMSNorm + RoPE on q/k planes (in place)
  rmsnorm_rope_kernel<<<dim3((2 * B_ * H_ * N_) / 4), 128, 0, stream>>>(
      qkv3, cosp, sinp, gq, gk);

  // flash attention -> f16 [B,N,C]
  flash_attn_kernel<<<dim3(N_ / 128, B_ * H_), 256, 0, stream>>>(qkv3, attnH);

  // output projection -> fp32 d_out
  gemm_f16_wmma<1><<<dim3(C_ / 128, BN_ / 128), 256, 0, stream>>>(
      attnH, WoutT, d_out, BN_, C_, C_);
}
